// STFT_14388140441718
// MI455X (gfx1250) — matrix operand order repaired and checked by reading, not compiled, not run on hardware
//
#include <hip/hip_runtime.h>

typedef __attribute__((ext_vector_type(16))) _Float16     v16h;
typedef __attribute__((ext_vector_type(8)))  float        v8f;
typedef __attribute__((ext_vector_type(8)))  unsigned int v8u;

#define WL     256
#define HOP    128
#define NSAMP  30720
#define NW     239        // frames per channel: (30720-256)/128 + 1
#define NBIN   129        // rfft bins
#define MT     15         // 16-frame M-tiles per channel (14 full + 1 partial)
#define NCH    1472       // 64 batches * 23 channels
#define NPAD   144        // bins padded to 9 N-tiles of 16
#define LDS_STRIDE 258    // 256 + 2 halves pad -> <=2-way LDS bank conflicts

// ---------------------------------------------------------------------------
// Init: f16 DFT tables in d_ws.  tab[n*256+k] = cos(2*pi*k*n/256),
// tab[NPAD*WL + n*256+k] = sin(2*pi*k*n/256).  (sign of sin irrelevant: we
// only need |X|.)  Angle reduced exactly via (k*n) mod 256.
// ---------------------------------------------------------------------------
__global__ void dft_table_init(_Float16* __restrict__ tab) {
    int idx = blockIdx.x * blockDim.x + threadIdx.x;   // 0 .. 144*256-1
    int n = idx >> 8;
    int k = idx & 255;
    int m = (k * n) & 255;
    float ang = (float)m * 0.0245436926f;              // pi/128
    tab[idx]             = (_Float16)__cosf(ang);
    tab[NPAD * WL + idx] = (_Float16)__sinf(ang);
}

// ---------------------------------------------------------------------------
// Main: one workgroup (9 waves / 288 threads) = 16 frames x 144 padded bins.
// Wave w computes N-tile w via v_wmma_f32_16x16x32_f16, K looped over 256.
// ---------------------------------------------------------------------------
__global__ void __launch_bounds__(288)
stft_wmma(const float* __restrict__ sig,
          const _Float16* __restrict__ tab,
          float* __restrict__ out) {
    __shared__ _Float16 smem[16 * LDS_STRIDE];

    const int    tid   = threadIdx.x;
    const int    blk   = blockIdx.x;
    const int    mt    = blk % MT;
    const int    ch    = blk / MT;             // flat (batch,channel) 0..1471
    const int    f0    = mt * 16;              // first frame of this M-tile
    const size_t sbase = (size_t)ch * NSAMP;

    // Stage windowed fp16 frame tile: smem[m][k] = x[f0+m, k] * hann(k)
    for (int i = tid; i < 16 * WL; i += 288) {
        int   m = i >> 8;
        int   k = i & 255;
        float v = 0.f;
        if (f0 + m < NW)
            v = sig[sbase + (size_t)(f0 + m) * HOP + k];
        float w = 0.5f - 0.5f * __cosf((float)k * 0.0246399424f); // 2*pi/255
        smem[m * LDS_STRIDE + k] = (_Float16)(v * w);
    }
    __syncthreads();

    const int lane = tid & 31;
    const int wv   = tid >> 5;       // 0..8 -> N-tile index
    const int g    = lane >> 4;      // lane half-group
    const int mrow = lane & 15;      // A row within tile / B column
    const int n0   = wv * 16;
    const int bin  = n0 + mrow;

    const _Float16* bc = tab + (size_t)bin * WL;               // cos row
    const _Float16* bs = tab + (size_t)(NPAD + bin) * WL;      // sin row

    v8f accR = {};
    v8f accI = {};

    #pragma unroll
    for (int kc = 0; kc < 8; ++kc) {
        const int kb = kc * 32;
        v8u au, bcu, bsu;
        #pragma unroll
        for (int v = 0; v < 8; ++v) {
            // A 16x32 f16 layout: VGPR v holds K pair; lanes16-31 offset +8
            int kpA = ((v >> 2) << 4) | ((v & 3) << 1) | (g << 3);
            au[v]  = *(const unsigned int*)(&smem[mrow * LDS_STRIDE + kb + kpA]);
            // B 32x16 f16 layout: same K order as A (VGPR v holds K pair; lanes16-31 offset +8)
            int kpB = ((v >> 2) << 4) | ((v & 3) << 1) | (g << 3);
            bcu[v] = *(const unsigned int*)(bc + kb + kpB);
            bsu[v] = *(const unsigned int*)(bs + kb + kpB);
        }
        v16h a  = __builtin_bit_cast(v16h, au);
        v16h b0 = __builtin_bit_cast(v16h, bcu);
        v16h b1 = __builtin_bit_cast(v16h, bsu);
        accR = __builtin_amdgcn_wmma_f32_16x16x32_f16(
                   false, a, false, b0, (short)0, accR, false, false);
        accI = __builtin_amdgcn_wmma_f32_16x16x32_f16(
                   false, a, false, b1, (short)0, accI, false, false);
    }

    // Epilogue: log-magnitude.  C/D layout: VGPR r -> M = r + 8*g, N = lane&15
    if (bin < NBIN) {
        #pragma unroll
        for (int r = 0; r < 8; ++r) {
            int frame = f0 + r + g * 8;
            if (frame < NW) {
                float re  = accR[r];
                float im  = accI[r];
                float mag = __builtin_sqrtf(re * re + im * im);
                out[((size_t)ch * NW + frame) * NBIN + bin] =
                    __logf(mag + 1e-8f) * 10.0f;
            }
        }
    }
}

// ---------------------------------------------------------------------------
extern "C" void kernel_launch(void* const* d_in, const int* in_sizes, int n_in,
                              void* d_out, int out_size, void* d_ws, size_t ws_size,
                              hipStream_t stream) {
    (void)in_sizes; (void)n_in; (void)out_size; (void)ws_size;
    const float* sig = (const float*)d_in[0];
    float*       out = (float*)d_out;
    _Float16*    tab = (_Float16*)d_ws;   // needs 2*144*256*2 = 147456 bytes

    dft_table_init<<<(NPAD * WL) / 256, 256, 0, stream>>>(tab);
    stft_wmma<<<NCH * MT, 288, 0, stream>>>(sig, tab, out);
}
